// S2Convolution_81020263071808
// MI455X (gfx1250) — compile-verified, matrix operand order checked by reading, not run on hardware
//
#include <hip/hip_runtime.h>
#include <hip/hip_bf16.h>
#include <math.h>
#include <stdint.h>

// ---------------------------------------------------------------------------
// S2 convolution (spherical CNN layer) for MI455X / gfx1250, wave32 + WMMA.
//   - all contractions on v_wmma_f32_16x16x32_f16 (f32 accumulate)
//   - reused operand matrices stored as _Float16 (no cvt in hot loops)
//   - twiddle tables staged to LDS via TDM (tensor_load_to_lds + tensorcnt)
//   - x tiles staged to LDS via global_load_async_to_lds_b32 (+ asynccnt)
//   - SO(3) spectrum (~44MB) stays L2-resident; dense 262MB grid never built
// ---------------------------------------------------------------------------

typedef __attribute__((ext_vector_type(16))) _Float16 v16h;
typedef __attribute__((ext_vector_type(8)))  float    v8f;
typedef unsigned int u32x4 __attribute__((ext_vector_type(4)));
typedef int          i32x8 __attribute__((ext_vector_type(8)));
typedef int          i32x4 __attribute__((ext_vector_type(4)));

#define PI_D 3.14159265358979323846
#define SCALING_F 8.16496580927726e-3   // 1/sqrt(24*16*20^4/64^2) = 1/sqrt(15000)

__host__ __device__ constexpr int S2sum(int l) { return l * (4 * l * l - 1) / 3; }

// ---- workspace layout (float units; "h" regions hold _Float16 pairs) -------
constexpr size_t OFF_WH   = 0;                         // half W[400][128]
constexpr size_t OFF_TRIG = OFF_WH   + 25600;          // half CT[48][132], ST[48][132]
constexpr size_t OFF_EH   = OFF_TRIG + 6336;           // half ER[48][64], EI[48][64]
constexpr size_t OFF_DK   = OFF_EH   + 3072;           // float d^l_{m,0}(grid) [400][3]
constexpr size_t OFF_YH   = OFF_DK   + 1200;           // half YR[400][512], YI[400][512]
constexpr size_t OFF_DI   = OFF_YH   + 204800;         // float d_inv flat (426400)
constexpr size_t OFF_XFR  = OFF_DI   + 426400;         // float [39][128][256]
constexpr size_t OFF_XFI  = OFF_XFR  + 39 * 128 * 256;
constexpr size_t OFF_XLH  = OFF_XFI  + 39 * 128 * 256; // half XLR[400][256], XLI[400][256]
constexpr size_t OFF_ZLR  = OFF_XLH  + 102400;         // float [10660][512]
constexpr size_t OFF_ZLI  = OFF_ZLR  + (size_t)10660 * 512;

// ---- WMMA helpers ----------------------------------------------------------
__device__ __forceinline__ v8f wmma32(v16h a, v16h b, v8f c) {
  return __builtin_amdgcn_wmma_f32_16x16x32_f16(false, a, false, b, (short)0, c,
                                                false, false);
}

// A fragment from half source: A[m][k] row-major, lane m = lane&15.
__device__ __forceinline__ v16h frag_a_h(const _Float16* p, int ld, int kbase,
                                         int klim, bool neg) {
  int lane = threadIdx.x & 31;
  int m = lane & 15, h = lane >> 4;
  v16h a;
#pragma unroll
  for (int t = 0; t < 8; ++t) {
    int k0 = kbase + (t >> 2) * 16 + h * 8 + 2 * (t & 3);
    _Float16 v0 = (k0 < klim) ? p[m * ld + k0] : (_Float16)0.f;
    _Float16 v1 = (k0 + 1 < klim) ? p[m * ld + k0 + 1] : (_Float16)0.f;
    a[2 * t]     = neg ? (_Float16)(-v0) : v0;
    a[2 * t + 1] = neg ? (_Float16)(-v1) : v1;
  }
  return a;
}
// B fragment from half source: B[k][n] row-major.
__device__ __forceinline__ v16h frag_b_h(const _Float16* p, int ld, int kbase, int klim) {
  int lane = threadIdx.x & 31;
  int n = lane & 15, h = lane >> 4;
  v16h b;
#pragma unroll
  for (int t = 0; t < 8; ++t) {
    int k0 = kbase + (t >> 2) * 16 + h * 8 + 2 * (t & 3);
    b[2 * t]     = (k0 < klim) ? p[k0 * ld + n] : (_Float16)0.f;
    b[2 * t + 1] = (k0 + 1 < klim) ? p[(k0 + 1) * ld + n] : (_Float16)0.f;
  }
  return b;
}
// B fragment, transposed half source: B[k][n] = p[n*ld + k].
__device__ __forceinline__ v16h frag_bT_h(const _Float16* p, int ld, int kbase, int klim) {
  int lane = threadIdx.x & 31;
  int n = lane & 15, h = lane >> 4;
  v16h b;
#pragma unroll
  for (int t = 0; t < 8; ++t) {
    int k0 = kbase + (t >> 2) * 16 + h * 8 + 2 * (t & 3);
    b[2 * t]     = (k0 < klim) ? p[n * ld + k0] : (_Float16)0.f;
    b[2 * t + 1] = (k0 + 1 < klim) ? p[n * ld + k0 + 1] : (_Float16)0.f;
  }
  return b;
}
// B fragment, transposed float source (x tiles): B[k][n] = p[n*ld + k].
__device__ __forceinline__ v16h frag_bT_f(const float* p, int ld, int kbase, int klim) {
  int lane = threadIdx.x & 31;
  int n = lane & 15, h = lane >> 4;
  v16h b;
#pragma unroll
  for (int t = 0; t < 8; ++t) {
    int k0 = kbase + (t >> 2) * 16 + h * 8 + 2 * (t & 3);
    b[2 * t]     = (_Float16)((k0 < klim) ? p[n * ld + k0] : 0.f);
    b[2 * t + 1] = (_Float16)((k0 + 1 < klim) ? p[n * ld + k0 + 1] : 0.f);
  }
  return b;
}

// ---- TDM: 1-D copy of `nh` halves from global to LDS (wave-uniform) --------
__device__ __forceinline__ void tdm_copy_to_lds(const void* gsrc, unsigned lds_addr,
                                                unsigned nh) {
  unsigned long long ga = (unsigned long long)(uintptr_t)gsrc;
  u32x4 g0;
  g0.x = 1u;                                              // count=1, user mode
  g0.y = lds_addr;                                        // lds_addr [63:32]
  g0.z = (unsigned)ga;                                    // global_addr[31:0]
  g0.w = (unsigned)((ga >> 32) & 0x01ffffffu) | 0x80000000u; // addr[56:32] | type=2
  i32x8 g1;
  g1[0] = (1 << 16);                                      // data_size=1 (2 bytes)
  g1[1] = (int)((nh & 0xffffu) << 16);                    // tensor_dim0[15:0]
  g1[2] = (int)((nh >> 16) & 0xffffu) | (1 << 16);        // dim0 hi | tensor_dim1=1
  g1[3] = (int)(nh << 16);                                // tile_dim0 (16b)
  g1[4] = 1;                                              // tile_dim1=1
  g1[5] = (int)nh;                                        // tensor_dim0_stride
  g1[6] = 0;
  g1[7] = 0;
  i32x4 z4 = {0, 0, 0, 0};
  i32x8 z8 = {0, 0, 0, 0, 0, 0, 0, 0};
  __builtin_amdgcn_tensor_load_to_lds(g0, g1, z4, z4, z8, 0);
}

// ---- Wigner small-d via log-factorial sum (f64) ----------------------------
__device__ double wig_d(int l, int m, int n, double beta, const double* lg) {
  double cb = cos(0.5 * beta), sb = sin(0.5 * beta);
  double lcb = log(fmax(cb, 1e-300)), lsb = log(fmax(sb, 1e-300));
  double pref = 0.5 * (lg[l + m] + lg[l - m] + lg[l + n] + lg[l - n]);
  int s0 = (n - m) > 0 ? (n - m) : 0;
  int s1 = (l + n) < (l - m) ? (l + n) : (l - m);
  double val = 0.0;
  for (int s = s0; s <= s1; ++s) {
    double c = pref - (lg[l + n - s] + lg[s] + lg[m - n + s] + lg[l - m - s]);
    double e = c + (double)(2 * l + n - m - 2 * s) * lcb +
               (double)(m - n + 2 * s) * lsb;
    double term = exp(e);
    val += (((m - n + s) & 1) ? -term : term);
  }
  return val;
}

__device__ __forceinline__ void build_lg(double* lg) {
  if (threadIdx.x == 0) {
    lg[0] = 0.0; lg[1] = 0.0;
    for (int k = 2; k < 48; ++k) lg[k] = lg[k - 1] + log((double)k);
  }
  __syncthreads();
}

__device__ __forceinline__ void decode_lm(int lm, int& l, int& m) {
  l = (int)sqrtf((float)lm + 0.5f);
  while (l * l > lm) --l;
  while ((l + 1) * (l + 1) <= lm) ++l;
  m = lm - l * l - l;
}

// ---- constant generation ----------------------------------------------------
__global__ void k_ws2(float* ws) {               // <<<400,128>>>
  __shared__ double lg[48];
  build_lg(lg);
  int lm = blockIdx.x, b = threadIdx.x;
  int l, m; decode_lm(lm, l, m);
  double t = PI_D * (2 * b + 1) / 256.0;
  double s = 0.0;
  for (int k = 0; k < 64; ++k) s += sin((2 * k + 1) * t) / (double)(2 * k + 1);
  double wq = (2.0 / 64.0) * sin(t) * s;
  _Float16* Wh = (_Float16*)(ws + OFF_WH);
  Wh[(size_t)lm * 128 + b] = (_Float16)(wig_d(l, m, 0, t, lg) * wq);
}

__global__ void k_trig(float* ws) {              // <<<25,256>>>
  int idx = blockIdx.x * 256 + threadIdx.x;
  _Float16* CTh = (_Float16*)(ws + OFF_TRIG);
  _Float16* STh = CTh + 48 * 132;
  _Float16* ERh = (_Float16*)(ws + OFF_EH);
  _Float16* EIh = ERh + 48 * 64;
  if (idx < 48 * 132) {
    int mi = idx / 132, a = idx % 132;
    double c = 0.0, s = 0.0;
    if (mi < 39 && a < 128) {
      double th = 2.0 * PI_D * (mi - 19) * a / 128.0;
      c = cos(th); s = -sin(th);
    }
    CTh[idx] = (_Float16)c;
    STh[idx] = (_Float16)s;
  }
  if (idx < 48 * 64) {
    int j = idx >> 6, mi = idx & 63;
    double c = 0.0, s = 0.0;
    if (j < 40 && mi < 39) {
      double th = 2.0 * PI_D * (mi - 19) * j / 40.0;
      c = cos(th); s = sin(th);
    }
    ERh[idx] = (_Float16)c;
    EIh[idx] = (_Float16)s;
  }
}

__global__ void k_dk(float* ws) {                // <<<5,256>>>
  __shared__ double lg[48];
  build_lg(lg);
  int idx = blockIdx.x * 256 + threadIdx.x;
  if (idx < 1200) {
    int lm = idx / 3, bi = idx % 3;
    int l, m; decode_lm(lm, l, m);
    double beta = (bi + 1) * (PI_D / 8.0) / 3.0;
    ws[OFF_DK + idx] = (float)wig_d(l, m, 0, beta, lg);
  }
}

__global__ void k_ybar(const float* __restrict__ ker, float* ws) {  // <<<400,512>>>
  int lm = blockIdx.x, io = threadIdx.x;
  int l, m; decode_lm(lm, l, m);
  double yr = 0.0, yi = 0.0;
  for (int p = 0; p < 24; ++p) {
    double kv = (double)ker[io * 24 + p];
    double dk = (double)ws[OFF_DK + lm * 3 + p / 8];
    double al = PI_D * (p & 7) / 4.0;        // conj(F_k) -> e^{+i m alpha}
    yr += kv * dk * cos(m * al);
    yi += kv * dk * sin(m * al);
  }
  _Float16* YRh = (_Float16*)(ws + OFF_YH);
  _Float16* YIh = YRh + (size_t)400 * 512;
  YRh[(size_t)lm * 512 + io] = (_Float16)(yr * SCALING_F);
  YIh[(size_t)lm * 512 + io] = (_Float16)(yi * SCALING_F);
}

__global__ void k_dinv(float* ws) {              // <<<dim3(238,20),256>>>
  __shared__ double lg[48];
  build_lg(lg);
  int l = blockIdx.y, n21 = 2 * l + 1;
  int tot = 40 * n21 * n21;
  int idx = blockIdx.x * 256 + threadIdx.x;
  if (idx < tot) {
    int b = idx / (n21 * n21), r = idx % (n21 * n21);
    int ml = r / n21, nl = r % n21;
    double beta = PI_D * (2 * b + 1) / 80.0;
    ws[OFF_DI + (size_t)40 * S2sum(l) + idx] =
        (float)((double)n21 * wig_d(l, ml - l, nl - l, beta, lg));
  }
}

// ---- stage 1: truncated DFT over alpha -------------------------------------
__global__ void __launch_bounds__(96) k_dft(const float* __restrict__ x, float* ws) {
  __shared__ _Float16 Trig[2 * 48 * 132];  // CT then ST, ld 132
  __shared__ float Xs[16 * 132];
  int tid = threadIdx.x;
  int zib = blockIdx.x * 16, b = blockIdx.y;
  __builtin_prefetch(x + (size_t)zib * 16384 + (size_t)b * 128, 0, 1);
  // TDM: DMA both twiddle tables (25.3KB) into LDS, wave 0 only.
  if (tid < 32) {
    tdm_copy_to_lds(ws + OFF_TRIG, (unsigned)(uintptr_t)&Trig[0], 2 * 48 * 132);
    __builtin_amdgcn_s_wait_tensorcnt(0);
  }
  // Async per-lane copy of the x tile into LDS.
  for (int i = tid; i < 16 * 128; i += 96) {
    int r = i >> 7, c = i & 127;
    unsigned lds = (unsigned)(uintptr_t)&Xs[r * 132 + c];
    unsigned long long ga =
        (unsigned long long)(uintptr_t)(x + (size_t)(zib + r) * 16384 +
                                        (size_t)b * 128 + c);
    asm volatile("global_load_async_to_lds_b32 %0, %1, off"
                 :: "v"(lds), "v"(ga) : "memory");
  }
  asm volatile("s_wait_asynccnt 0" ::: "memory");
  __syncthreads();

  const _Float16* Cs = Trig;
  const _Float16* Ss = Trig + 48 * 132;
  int w = tid >> 5, lane = tid & 31, n = lane & 15, h = lane >> 4;
  v8f ar = {}, ai = {};
#pragma unroll
  for (int kk = 0; kk < 128; kk += 32) {
    v16h bf = frag_bT_f(Xs, 132, kk, 128);               // B[a][zi] = Xs[zi][a]
    v16h ac = frag_a_h(Cs + w * 16 * 132, 132, kk, 128, false);
    v16h as = frag_a_h(Ss + w * 16 * 132, 132, kk, 128, false);
    ar = wmma32(ac, bf, ar);
    ai = wmma32(as, bf, ai);
  }
#pragma unroll
  for (int vr = 0; vr < 8; ++vr) {
    int mi = w * 16 + vr + 8 * h;
    if (mi < 39) {
      size_t o = ((size_t)mi * 128 + b) * 256 + zib + n;
      ws[OFF_XFR + o] = ar[vr];
      ws[OFF_XFI + o] = ai[vr];
    }
  }
}

// ---- stage 2: S2 beta analysis per m ---------------------------------------
__global__ void __launch_bounds__(64) k_s2(float* ws) {  // <<<dim3(16,39),64>>>
  __shared__ _Float16 As[32 * 132], Bsr[128 * 16], Bsi[128 * 16];
  int tid = threadIdx.x;
  int mi = blockIdx.y, zib = blockIdx.x * 16;
  int m = mi - 19, am = m < 0 ? -m : m, Lm = 20 - am;
  const _Float16* Wh = (const _Float16*)(ws + OFF_WH);
  for (int i = tid; i < 32 * 132; i += 64) {
    int lp = i / 132, bb = i % 132;
    _Float16 v = (_Float16)0.f;
    if (lp < Lm && bb < 128) {
      int l = am + lp;
      v = Wh[(size_t)(l * l + m + l) * 128 + bb];
    }
    As[i] = v;
  }
  for (int i = tid; i < 128 * 16; i += 64) {
    int bb = i >> 4, nn = i & 15;
    size_t o = ((size_t)mi * 128 + bb) * 256 + zib + nn;
    Bsr[i] = (_Float16)ws[OFF_XFR + o];
    Bsi[i] = (_Float16)ws[OFF_XFI + o];
  }
  __syncthreads();
  int w = tid >> 5, lane = tid & 31, n = lane & 15, h = lane >> 4;
  _Float16* XLRh = (_Float16*)(ws + OFF_XLH);
  _Float16* XLIh = XLRh + (size_t)400 * 256;
  if (w * 16 < Lm) {                                      // wave-uniform
    v8f xr = {}, xi = {};
#pragma unroll
    for (int kk = 0; kk < 128; kk += 32) {
      v16h a = frag_a_h(As + w * 16 * 132, 132, kk, 128, false);
      v16h br = frag_b_h(Bsr, 16, kk, 128);
      v16h bi = frag_b_h(Bsi, 16, kk, 128);
      xr = wmma32(a, br, xr);
      xi = wmma32(a, bi, xi);
    }
#pragma unroll
    for (int vr = 0; vr < 8; ++vr) {
      int lp = w * 16 + vr + 8 * h;
      if (lp < Lm) {
        int l = am + lp;
        size_t o = (size_t)(l * l + m + l) * 256 + zib + n;
        XLRh[o] = (_Float16)xr[vr];
        XLIh[o] = (_Float16)xi[vr];
      }
    }
  }
}

// ---- stage 3: SO(3) spectrum zl[m,n,z,o] (complex, K=16) -------------------
__global__ void __launch_bounds__(32) k_so3(float* ws) {  // <<<dim3(78,39,20),32>>>
  int l = blockIdx.z, n21 = 2 * l + 1;
  int m = blockIdx.y; if (m >= n21) return;
  int ct = blockIdx.x; if (ct >= 2 * n21) return;
  int nn = ct >> 1, oh = ct & 1;
  const _Float16* XLRh = (const _Float16*)(ws + OFF_XLH);
  const _Float16* XLIh = XLRh + (size_t)400 * 256;
  const _Float16* YRh = (const _Float16*)(ws + OFF_YH);
  const _Float16* YIh = YRh + (size_t)400 * 512;
  const _Float16* Ar = XLRh + (size_t)(l * l + m) * 256;        // A[z][i], ld16
  const _Float16* Ai = XLIh + (size_t)(l * l + m) * 256;
  const _Float16* Br = YRh + (size_t)(l * l + nn) * 512 + oh * 16; // B[i][o], ld32
  const _Float16* Bi = YIh + (size_t)(l * l + nn) * 512 + oh * 16;
  v16h ar  = frag_a_h(Ar, 16, 0, 16, false);
  v16h aip = frag_a_h(Ai, 16, 0, 16, false);
  v16h ain = frag_a_h(Ai, 16, 0, 16, true);
  v16h br  = frag_b_h(Br, 32, 0, 16);
  v16h bi  = frag_b_h(Bi, 32, 0, 16);
  v8f z = {};
  v8f dr = wmma32(ar, br, z);  dr = wmma32(ain, bi, dr);   // Re = xr*yr - xi*yi
  v8f di = wmma32(ar, bi, z);  di = wmma32(aip, br, di);   // Im = xr*yi + xi*yr
  int lane = threadIdx.x & 31, oo = lane & 15, h = lane >> 4;
  size_t base = (size_t)512 * S2sum(l) + (size_t)(m * n21 + nn) * 512 + oh * 16 + oo;
#pragma unroll
  for (int vr = 0; vr < 8; ++vr) {
    int zz = vr + 8 * h;
    ws[OFF_ZLR + base + (size_t)zz * 32] = dr[vr];
    ws[OFF_ZLI + base + (size_t)zz * 32] = di[vr];
  }
}

// ---- stage 4: per (zo, b) gather + fused inverse 2D transform --------------
__global__ void __launch_bounds__(96) k_synth(const float* __restrict__ bias,
                                              float* __restrict__ out,
                                              const float* __restrict__ ws) {
  __shared__ _Float16 Mr[64 * 48], Mi[64 * 48];   // [mi(pad64)][ni(pad48)]
  __shared__ _Float16 Tr[48 * 68], Ti[48 * 68];   // [j(pad48)][n(pad64, ld68)]
  int tid = threadIdx.x;
  int b = blockIdx.x, zo = blockIdx.y;
  for (int i = tid; i < 64 * 48; i += 96) { Mr[i] = (_Float16)0.f; Mi[i] = (_Float16)0.f; }
  __syncthreads();
  // gather M[mi][ni] = sum_l zl[l][m][n][zo] * d_inv[l][b][m][n]
  for (int p = tid; p < 39 * 39; p += 96) {
    int mi = p / 39, ni = p % 39;
    int am = mi < 19 ? 19 - mi : mi - 19;
    int an = ni < 19 ? 19 - ni : ni - 19;
    int lmin = am > an ? am : an;
    float accr = 0.f, acci = 0.f;
    for (int l = lmin; l < 20; ++l) {
      int n21 = 2 * l + 1;
      int r = (mi - 19 + l) * n21 + (ni - 19 + l);
      size_t zb = (size_t)512 * S2sum(l) + (size_t)r * 512 + zo;
      float d = ws[OFF_DI + (size_t)40 * S2sum(l) + (size_t)b * n21 * n21 + r];
      accr += ws[OFF_ZLR + zb] * d;
      acci += ws[OFF_ZLI + zb] * d;
    }
    Mr[mi * 48 + ni] = (_Float16)accr;
    Mi[mi * 48 + ni] = (_Float16)acci;
  }
  __syncthreads();
  const _Float16* ERh = (const _Float16*)(ws + OFF_EH);
  const _Float16* EIh = ERh + 48 * 64;
  int w = tid >> 5, lane = tid & 31, n = lane & 15, h = lane >> 4;
  // step 1: T = E * M   (complex; E is [48][64] padded)
  for (int ct = 0; ct < 3; ++ct) {
    v8f tr = {}, ti = {};
#pragma unroll
    for (int kk = 0; kk < 64; kk += 32) {
      v16h aer  = frag_a_h(ERh + w * 16 * 64, 64, kk, 64, false);
      v16h aei  = frag_a_h(EIh + w * 16 * 64, 64, kk, 64, false);
      v16h aein = frag_a_h(EIh + w * 16 * 64, 64, kk, 64, true);
      v16h bmr  = frag_b_h(Mr + ct * 16, 48, kk, 64);
      v16h bmi  = frag_b_h(Mi + ct * 16, 48, kk, 64);
      tr = wmma32(aer, bmr, tr);  tr = wmma32(aein, bmi, tr);
      ti = wmma32(aer, bmi, ti);  ti = wmma32(aei, bmr, ti);
    }
#pragma unroll
    for (int vr = 0; vr < 8; ++vr) {
      int j = w * 16 + vr + 8 * h;
      Tr[j * 68 + ct * 16 + n] = (_Float16)tr[vr];
      Ti[j * 68 + ct * 16 + n] = (_Float16)ti[vr];
    }
  }
  __syncthreads();
  // step 2: F = Re(T * E^T) + bias, store final output
  float bv = bias[zo & 31];
  for (int ctk = 0; ctk < 3; ++ctk) {
    v8f f = {};
#pragma unroll
    for (int kk = 0; kk < 64; kk += 32) {
      v16h atr  = frag_a_h(Tr + w * 16 * 68, 68, kk, 48, false);
      v16h atin = frag_a_h(Ti + w * 16 * 68, 68, kk, 48, true);
      v16h ber  = frag_bT_h(ERh + ctk * 16 * 64, 64, kk, 64);
      v16h bei  = frag_bT_h(EIh + ctk * 16 * 64, 64, kk, 64);
      f = wmma32(atr, ber, f);
      f = wmma32(atin, bei, f);
    }
#pragma unroll
    for (int vr = 0; vr < 8; ++vr) {
      int j = w * 16 + vr + 8 * h, k = ctk * 16 + n;
      if (j < 40 && k < 40)
        out[((size_t)zo * 40 + b) * 1600 + j * 40 + k] = f[vr] + bv;
    }
  }
}

// ---------------------------------------------------------------------------
extern "C" void kernel_launch(void* const* d_in, const int* in_sizes, int n_in,
                              void* d_out, int out_size, void* d_ws, size_t ws_size,
                              hipStream_t stream) {
  (void)in_sizes; (void)n_in; (void)out_size; (void)ws_size;
  const float* x    = (const float*)d_in[0];   // [16,16,128,128]
  const float* ker  = (const float*)d_in[1];   // [16,32,24]
  const float* bias = (const float*)d_in[2];   // [32]
  float* out = (float*)d_out;                  // [16,32,40,40,40]
  float* ws  = (float*)d_ws;                   // ~57 MB used

  k_ws2 <<<400, 128, 0, stream>>>(ws);
  k_trig<<<25, 256, 0, stream>>>(ws);
  k_dk  <<<5, 256, 0, stream>>>(ws);
  k_ybar<<<400, 512, 0, stream>>>(ker, ws);
  k_dinv<<<dim3(238, 20), 256, 0, stream>>>(ws);

  k_dft <<<dim3(16, 128), 96, 0, stream>>>(x, ws);
  k_s2  <<<dim3(16, 39), 64, 0, stream>>>(ws);
  k_so3 <<<dim3(78, 39, 20), 32, 0, stream>>>(ws);
  k_synth<<<dim3(40, 512), 96, 0, stream>>>(bias, out, ws);
}
